// TorchFuseAllTiled_58067957842552
// MI455X (gfx1250) — matrix operand order repaired and checked by reading, not compiled, not run on hardware
//
#include <hip/hip_runtime.h>
#include <cstdint>

#define TOKENS 4096
#define DMODEL 512
#define VOCAB  128000

#define M_TILE 32
#define N_TILE 512
#define KSTEP  32
#define NVBLK (VOCAB / N_TILE)   /* 250 vocab blocks */
#define NTBLK (TOKENS / M_TILE)  /* 128 token blocks */

/* fp32 LDS layouts, padded so 16 lanes hit distinct bank groups for b128 */
#define A_STRIDEF 516                              /* floats per A row (512+4) */
#define A_BYTES   (M_TILE * A_STRIDEF * 4)         /* 66048 */
#define B_STRIDEF 36                               /* floats per B row (32+4)  */
#define BSTAGE_BYTES (N_TILE * B_STRIDEF * 4)      /* 73728 per buffer */
#define STATS_OFF (A_BYTES + 2 * BSTAGE_BYTES)
#define SMEM_BYTES (STATS_OFF + M_TILE * 8 * 4)    /* ~210 KB < 320 KB WGP LDS */

typedef __attribute__((ext_vector_type(16))) _Float16 v16h;
typedef __attribute__((ext_vector_type(2)))  __fp16   v2hf;
typedef __attribute__((ext_vector_type(16))) __fp16   v16hf;
typedef __attribute__((ext_vector_type(8)))  float    v8f;

// compiler memory fence: LDS loads issued above cannot sink below this point
#define SCHED_FENCE() asm volatile("" ::: "memory")

// ---------------------------------------------------------------------------
// CDNA5 async copy: global -> LDS, 16 bytes per lane, tracked by ASYNCcnt.
// GVS addressing: 64-bit SGPR base + 32-bit per-lane VGPR offset.
// ---------------------------------------------------------------------------
__device__ __forceinline__ void async_b128(uint32_t lds_addr, uint32_t voff,
                                           unsigned long long base) {
  asm volatile("global_load_async_to_lds_b128 %0, %1, %2"
               :: "v"(lds_addr), "v"(voff), "s"(base)
               : "memory");
}
__device__ __forceinline__ void wait_async0() {
  asm volatile("s_wait_asynccnt 0x0" ::: "memory");
}

// pack 16 fp32 -> v16h with v_cvt_pk_rtz_f16_f32 (co-executes with XDL WMMA)
__device__ __forceinline__ v16h cvt16(const float4* f) {
  v16hf b;
#pragma unroll
  for (int q = 0; q < 4; ++q) {
    v2hf p0 = __builtin_amdgcn_cvt_pkrtz(f[q].x, f[q].y);
    v2hf p1 = __builtin_amdgcn_cvt_pkrtz(f[q].z, f[q].w);
    b[q * 4 + 0] = p0[0]; b[q * 4 + 1] = p0[1];
    b[q * 4 + 2] = p1[0]; b[q * 4 + 3] = p1[1];
  }
  return __builtin_bit_cast(v16h, b);
}

__device__ __forceinline__ void ldB16(float4* dst, const float* bp) {
  dst[0] = *(const float4*)(bp + 0);
  dst[1] = *(const float4*)(bp + 4);
  dst[2] = *(const float4*)(bp + 8);
  dst[3] = *(const float4*)(bp + 12);
}

// ---------------------------------------------------------------------------
// Kernel 1: WMMA GEMM tile (M_TILE x N_TILE, K=512) fused with per-row
// max / sum-exp.  Block = 256 threads = 8 waves arranged 2 (M) x 4 (N).
// LDS (210 KB) limits residency to ONE workgroup per WGP => exactly 2
// waves/SIMD; amdgpu_waves_per_eu(2,2) tells the allocator so it stops
// trading registers for unreachable occupancy.
// Data path: async-DMA fp32 tiles into LDS (double-buffered for W), convert
// to f16 at fragment-load time, v_wmma_f32_16x16x32_f16 with fp32 accum.
// ---------------------------------------------------------------------------
__global__ __launch_bounds__(256)
__attribute__((amdgpu_waves_per_eu(2, 2)))
void ce_partials_kernel(
    const float* __restrict__ x, const float* __restrict__ w,
    float* __restrict__ pmax, float* __restrict__ psum)
{
  extern __shared__ char smem[];
  const float* As = (const float*)smem;              // [M_TILE][A_STRIDEF] fp32
  float* St = (float*)(smem + STATS_OFF);            // [M_TILE][4][2]

  const int tid  = threadIdx.x;
  const int lane = tid & 31;
  const int wid  = tid >> 5;
  const int wm   = wid >> 2;     // wave row    0..1 (16 tokens each)
  const int wn   = wid & 3;      // wave column 0..3 (128 vocab each)
  const int l15  = lane & 15;
  const int hsel = lane >> 4;    // which K-half this lane holds

  const int t0 = blockIdx.x * M_TILE;   // token base
  const int v0 = blockIdx.y * N_TILE;   // vocab base

  const uint32_t lds0 = __builtin_amdgcn_groupstaticsize();  // dynamic-LDS base

  // ---- prologue: async-stage A (whole x tile, fp32) and B chunk for ko=0 --
  {
    const unsigned long long abase =
        (unsigned long long)(uintptr_t)(x + (size_t)t0 * DMODEL);
#pragma unroll
    for (int i = tid * 4; i < M_TILE * DMODEL; i += 256 * 4) {
      const int m = i >> 9, k = i & (DMODEL - 1);
      async_b128(lds0 + (uint32_t)(m * (A_STRIDEF * 4) + k * 4),
                 (uint32_t)((m * DMODEL + k) * 4), abase);
    }
    const unsigned long long bbase =
        (unsigned long long)(uintptr_t)(w + (size_t)v0 * DMODEL);
#pragma unroll
    for (int j = tid * 4; j < N_TILE * KSTEP; j += 256 * 4) {
      const int n = j >> 5, k = j & 31;
      async_b128(lds0 + (uint32_t)(A_BYTES + n * (B_STRIDEF * 4) + k * 4),
                 (uint32_t)((n * DMODEL + k) * 4), bbase);
    }
  }

  v8f acc[8];
  const v8f vzero = {};
#pragma unroll
  for (int t = 0; t < 8; ++t) acc[t] = vzero;

  // ---- main K loop: async(ko+1) overlaps WMMA(ko) -------------------------
  for (int ko = 0; ko < DMODEL / KSTEP; ++ko) {
    const int k0 = ko * KSTEP;
    wait_async0();        // this wave's async copies for step ko have landed
    __syncthreads();      // ... and everyone else's too

    if (ko + 1 < DMODEL / KSTEP) {   // kick off W slice for the next step
      const unsigned long long bbase = (unsigned long long)(uintptr_t)(
          w + (size_t)v0 * DMODEL + (k0 + KSTEP));
      const uint32_t buf = A_BYTES + (uint32_t)((ko + 1) & 1) * BSTAGE_BYTES;
#pragma unroll
      for (int j = tid * 4; j < N_TILE * KSTEP; j += 256 * 4) {
        const int n = j >> 5, k = j & 31;
        async_b128(lds0 + buf + (uint32_t)(n * (B_STRIDEF * 4) + k * 4),
                   (uint32_t)((n * DMODEL + k) * 4), bbase);
      }
    }

    const float* Bcur = (const float*)(smem + A_BYTES
                                       + (size_t)(ko & 1) * BSTAGE_BYTES);

    // A fragment (16x32): lanes<16 hold K {0-15}, lanes>=16 {16-31}
    const float* ap = As + (wm * 16 + l15) * A_STRIDEF + k0 + hsel * 16;
    float4 af[4];
    af[0] = *(const float4*)(ap + 0);
    af[1] = *(const float4*)(ap + 4);
    af[2] = *(const float4*)(ap + 8);
    af[3] = *(const float4*)(ap + 12);
    const v16h a = cvt16(af);

    // B fragments: 4 groups of 2 subtiles, two distinct register buffers,
    // fences pin each group's 8 ds_loads ahead of the previous group's math.
    // Within a group: convert BOTH fragments, then issue both WMMAs
    // back-to-back (2nd WMMA covers the 1st's WAR hazard window).
    const float* bbase_lds = Bcur + (wn * 128 + l15) * B_STRIDEF + hsel * 16;
    float4 bufA[8], bufB[8];
    ldB16(&bufA[0], bbase_lds + 0 * (16 * B_STRIDEF));
    ldB16(&bufA[4], bbase_lds + 1 * (16 * B_STRIDEF));
    SCHED_FENCE();
#pragma unroll
    for (int g = 0; g < 4; ++g) {
      const float4* cur = (g & 1) ? bufB : bufA;
      float4*       nxt = (g & 1) ? bufA : bufB;
      if (g < 3) {
        ldB16(&nxt[0], bbase_lds + ((g + 1) * 2 + 0) * (16 * B_STRIDEF));
        ldB16(&nxt[4], bbase_lds + ((g + 1) * 2 + 1) * (16 * B_STRIDEF));
        SCHED_FENCE();
      }
      const v16h b0 = cvt16(&cur[0]);
      const v16h b1 = cvt16(&cur[4]);
      acc[g * 2 + 0] = __builtin_amdgcn_wmma_f32_16x16x32_f16(
          false, a, false, b0, (short)0, acc[g * 2 + 0], false, false);
      acc[g * 2 + 1] = __builtin_amdgcn_wmma_f32_16x16x32_f16(
          false, a, false, b1, (short)0, acc[g * 2 + 1], false, false);
    }
  }

  // ---- epilogue: per-row max & sum-exp over this wave's 128 columns -------
  // C layout: VGPR r: lanes<16 -> (row r, col lane); lanes>=16 -> (row 8+r, col lane-16)
#pragma unroll
  for (int r = 0; r < 8; ++r) {
    float mx = -3.402823466e38f;
#pragma unroll
    for (int t = 0; t < 8; ++t) mx = fmaxf(mx, acc[t][r]);
#pragma unroll
    for (int off = 1; off < 16; off <<= 1)
      mx = fmaxf(mx, __shfl_xor(mx, off, 32));   // stays within 16-lane half
    float s = 0.f;
#pragma unroll
    for (int t = 0; t < 8; ++t) s += __expf(acc[t][r] - mx);
#pragma unroll
    for (int off = 1; off < 16; off <<= 1)
      s += __shfl_xor(s, off, 32);
    if (l15 == 0) {
      const int rl = wm * 16 + hsel * 8 + r;     // row within block tile
      St[rl * 8 + wn * 2 + 0] = mx;
      St[rl * 8 + wn * 2 + 1] = s;
    }
  }
  __syncthreads();

  // merge the four wave-columns -> one (max,sum) per token for this vocab block
  if (tid < M_TILE) {
    float M = St[tid * 8 + 0];
#pragma unroll
    for (int c = 1; c < 4; ++c) M = fmaxf(M, St[tid * 8 + c * 2]);
    float S = 0.f;
#pragma unroll
    for (int c = 0; c < 4; ++c)
      S += St[tid * 8 + c * 2 + 1] * __expf(St[tid * 8 + c * 2] - M);
    const size_t o = (size_t)blockIdx.y * TOKENS + t0 + tid;  // coalesced
    pmax[o] = M;
    psum[o] = S;
  }
}

// ---------------------------------------------------------------------------
// Kernel 2: target score  tscore[t] = dot(x[t], W[target[t]])  in fp32.
// ---------------------------------------------------------------------------
__global__ __launch_bounds__(256) void target_score_kernel(
    const float* __restrict__ x, const float* __restrict__ w,
    const int* __restrict__ target, float* __restrict__ tscore)
{
  const int lane = threadIdx.x & 31;
  const int wid  = threadIdx.x >> 5;
  const int t    = blockIdx.x * 8 + wid;
  const int tg   = target[t];
  const float* xr = x + (size_t)t  * DMODEL;
  const float* wr = w + (size_t)tg * DMODEL;
  float s = 0.f;
#pragma unroll
  for (int k = lane * 4; k < DMODEL; k += 128) {
    const float4 a = *(const float4*)(xr + k);
    const float4 b = *(const float4*)(wr + k);
    s = fmaf(a.x, b.x, fmaf(a.y, b.y, fmaf(a.z, b.z, fmaf(a.w, b.w, s))));
  }
#pragma unroll
  for (int off = 1; off < 32; off <<= 1) s += __shfl_xor(s, off, 32);
  if (lane == 0) tscore[t] = s;
}

// ---------------------------------------------------------------------------
// Kernel 3: merge the 250 per-vocab-block partials per token (online softmax
// combine), form loss term, reduce to scalar via atomicAdd.
// ---------------------------------------------------------------------------
__global__ __launch_bounds__(256) void finalize_kernel(
    const float* __restrict__ pmax, const float* __restrict__ psum,
    const float* __restrict__ tscore, float* __restrict__ out)
{
  __shared__ float wsum[8];
  const int lane = threadIdx.x & 31;
  const int wid  = threadIdx.x >> 5;
  const int t    = blockIdx.x * 8 + wid;

  float mx = -3.402823466e38f;
  for (int j = lane; j < NVBLK; j += 32)
    mx = fmaxf(mx, pmax[(size_t)j * TOKENS + t]);
#pragma unroll
  for (int off = 1; off < 32; off <<= 1) mx = fmaxf(mx, __shfl_xor(mx, off, 32));

  float s = 0.f;
  for (int j = lane; j < NVBLK; j += 32) {
    const size_t o = (size_t)j * TOKENS + t;
    s += psum[o] * __expf(pmax[o] - mx);
  }
#pragma unroll
  for (int off = 1; off < 32; off <<= 1) s += __shfl_xor(s, off, 32);

  if (lane == 0) wsum[wid] = (mx + __logf(s)) - tscore[t];
  __syncthreads();
  if (threadIdx.x == 0) {
    float a = 0.f;
#pragma unroll
    for (int i = 0; i < 8; ++i) a += wsum[i];
    atomicAdd(out, a);
  }
}

__global__ void zero_out_kernel(float* __restrict__ out) {
  if (threadIdx.x == 0 && blockIdx.x == 0) out[0] = 0.f;
}

// ---------------------------------------------------------------------------
extern "C" void kernel_launch(void* const* d_in, const int* in_sizes, int n_in,
                              void* d_out, int out_size, void* d_ws, size_t ws_size,
                              hipStream_t stream) {
  (void)in_sizes; (void)n_in; (void)out_size; (void)ws_size;
  const float* x      = (const float*)d_in[0];
  const float* w      = (const float*)d_in[1];
  const int*   target = (const int*)d_in[2];
  float* out = (float*)d_out;

  // workspace (~8.2 MB): pmax[NVBLK][TOKENS], psum[NVBLK][TOKENS], tscore[TOKENS]
  float* pmax   = (float*)d_ws;
  float* psum   = pmax + (size_t)NVBLK * TOKENS;
  float* tscore = psum + (size_t)NVBLK * TOKENS;

  (void)hipFuncSetAttribute(reinterpret_cast<const void*>(&ce_partials_kernel),
                            hipFuncAttributeMaxDynamicSharedMemorySize,
                            (int)SMEM_BYTES);

  zero_out_kernel<<<1, 64, 0, stream>>>(out);
  target_score_kernel<<<TOKENS / 8, 256, 0, stream>>>(x, w, target, tscore);
  dim3 grid(NTBLK, NVBLK);   // token-fast ordering -> W slice reused out of L2
  ce_partials_kernel<<<grid, 256, (size_t)SMEM_BYTES, stream>>>(x, w, pmax, psum);
  finalize_kernel<<<TOKENS / 8, 256, 0, stream>>>(pmax, psum, tscore, out);
}
